// DynRNN_18949395710025
// MI455X (gfx1250) — matrix operand order repaired and checked by reading, not compiled, not run on hardware
//
#include <hip/hip_runtime.h>

// LSTM: B=2048, T=16, F=768, H=1024.
// gates = [x_t | h] @ [W_ih; W_hh]^ + b, fused with cell update, bf16 WMMA.
// x and W pre-converted to bf16 once; h carried as bf16 between steps.
// Step kernel: double-buffered async global->LDS pipeline (ASYNCcnt).

#define B_N   2048
#define T_N   16
#define F_N   768
#define H_N   1024
#define K_N   (F_N + H_N)   // 1792
#define KCH   (K_N / 32)    // 56 chunks of K=32
#define M_BLK 256           // block tile: 256 batch rows x 16 h-cols
#define APC   5             // async copies issued per wave per chunk (4 A + 1 B)

typedef __attribute__((ext_vector_type(16))) __bf16 v16bf;
typedef __attribute__((ext_vector_type(8)))  __bf16 v8bf;
typedef __attribute__((ext_vector_type(8)))  float  v8f;
typedef __attribute__((ext_vector_type(4)))  int    v4i;

#define AS1 __attribute__((address_space(1)))
#define AS3 __attribute__((address_space(3)))

#if __has_builtin(__builtin_amdgcn_global_load_async_to_lds_b128)
#define HAVE_ASYNC 1
#else
#define HAVE_ASYNC 0
#endif

static __device__ __forceinline__ unsigned short f2bf(float x) {
  unsigned u = __float_as_uint(x);
  u += 0x7FFFu + ((u >> 16) & 1u);   // round-to-nearest-even
  return (unsigned short)(u >> 16);
}

static __device__ __forceinline__ float sigf(float x) {
  return 1.0f / (1.0f + __expf(-x));
}

static __device__ __forceinline__ float tanh_fast(float x) {
  float cx = fminf(fmaxf(x, -15.0f), 15.0f);
  float e  = __expf(2.0f * cx);
  return (e - 1.0f) / (e + 1.0f);
}

// 16-byte global -> LDS copy; async (no VGPR round trip) when available.
static __device__ __forceinline__ void cp16(const unsigned short* g,
                                            unsigned short* l) {
#if HAVE_ASYNC
  __builtin_amdgcn_global_load_async_to_lds_b128(
      (AS1 v4i*)g, (AS3 v4i*)l, 0, 0);
#else
  *(uint4*)l = *(const uint4*)g;
#endif
}

template <int N>
static __device__ __forceinline__ void wait_async() {
#if HAVE_ASYNC
#if __has_builtin(__builtin_amdgcn_s_wait_asynccnt)
  __builtin_amdgcn_s_wait_asynccnt(N);
#else
  asm volatile("s_wait_asynccnt %0" ::"i"(N) : "memory");
#endif
#endif
}

// One-time: WT[n][k] (bf16) = concat(W_ih, W_hh)^T ; 14.7 MB, L2-resident.
__global__ __launch_bounds__(256) void prep_wt_kernel(
    const float* __restrict__ Wih, const float* __restrict__ Whh,
    unsigned short* __restrict__ WT)
{
  size_t idx = (size_t)blockIdx.x * 256 + threadIdx.x;  // 4096*1792
  int n = (int)(idx / K_N);
  int k = (int)(idx % K_N);
  float v = (k < F_N) ? Wih[(size_t)k * (4 * H_N) + n]
                      : Whh[(size_t)(k - F_N) * (4 * H_N) + n];
  WT[idx] = f2bf(v);
}

// One-time: x -> bf16 (amortized over the 16 steps that re-read it).
__global__ __launch_bounds__(256) void prep_x_kernel(
    const float* __restrict__ x, unsigned short* __restrict__ xbf)
{
  size_t idx = (size_t)blockIdx.x * 256 + threadIdx.x;  // 2048*16*768
  xbf[idx] = f2bf(x[idx]);
}

__global__ __launch_bounds__(256) void init_state_kernel(
    unsigned short* __restrict__ h0, float* __restrict__ c)
{
  size_t idx = (size_t)blockIdx.x * 256 + threadIdx.x;  // 2048*1024
  h0[idx] = 0;
  c[idx]  = 0.0f;
}

// Block: 256 thr = 8 wave32. Tile: 256 batch rows x 16 h-cols.
// Wave: 32 rows x 16 cols -> 2 M-subtiles x 4 gates = 8 WMMA / K-chunk,
// each B fragment reused across both M-subtiles.
__global__ __launch_bounds__(256) void lstm_step_kernel(
    const unsigned short* __restrict__ xbf,
    const unsigned short* __restrict__ WT,
    const float*          __restrict__ bias,
    const unsigned short* __restrict__ hprev,
    unsigned short*       __restrict__ hnext,
    float*                __restrict__ cst,
    float*                __restrict__ outf,
    int t, int last)
{
  __shared__ __align__(16) unsigned short Alds[2][M_BLK][32];   // 2 x 16 KB
  __shared__ __align__(16) unsigned short Blds[2][4][16][32];   // 2 x  4 KB

  const int tid   = threadIdx.x;
  const int lane  = tid & 31;
  const int wave  = tid >> 5;
  const int mBase = blockIdx.x * M_BLK;   // 8 blocks over B
  const int n0    = blockIdx.y * 16;      // 64 blocks over H
  const int lidx  = lane & 15;
  const int half  = lane >> 4;

  v8f acc[2][4] = {};   // [m-subtile][gate]

  // Stage one K-chunk (32 wide) into LDS buffer `buf`.
  auto stage = [&](int kc, int buf) {
    const int k0 = kc * 32;
    // A: 256 rows x 32 K bf16 = 16 KB -> 4x b128 per thread
#pragma unroll
    for (int i = 0; i < 4; ++i) {
      int lin  = tid + i * 256;
      int row  = lin >> 2;
      int koff = (lin & 3) * 8;
      const unsigned short* g;
      if (k0 < F_N) {
        g = xbf + ((size_t)(mBase + row) * T_N + t) * F_N + k0 + koff;
      } else {
        g = hprev + (size_t)(mBase + row) * H_N + (k0 - F_N) + koff;
      }
      cp16(g, &Alds[buf][row][koff]);
    }
    // B: 4 gates x 16 n x 32 K bf16 = 4 KB -> 1x b128 per thread
    {
      int gg   = tid >> 6;
      int rr   = (tid >> 2) & 15;
      int koff = (tid & 3) * 8;
      cp16(WT + (size_t)(gg * H_N + n0 + rr) * K_N + k0 + koff,
           &Blds[buf][gg][rr][koff]);
    }
  };

  stage(0, 0);
  for (int kc = 0; kc < KCH; ++kc) {
    const int buf = kc & 1;
    if (kc + 1 < KCH) {
      stage(kc + 1, buf ^ 1);   // overlap next chunk's copies with wait
      wait_async<APC>();        // previous APC asyncs (this chunk) done
    } else {
      wait_async<0>();
    }
    __syncthreads();            // staged data visible to all waves

    // A fragments: lane<16 -> K {0..7,16..23}; lane>=16 -> K {8..15,24..31}
    v16bf aF[2];
#pragma unroll
    for (int ms = 0; ms < 2; ++ms) {
      const v8bf* ap =
          (const v8bf*)&Alds[buf][wave * 32 + ms * 16 + lidx][half * 8];
      aF[ms] = __builtin_shufflevector(ap[0], ap[2],
          0, 1, 2, 3, 4, 5, 6, 7, 8, 9, 10, 11, 12, 13, 14, 15);
    }
#pragma unroll
    for (int g = 0; g < 4; ++g) {
      // B fragment: column n0+lidx; lanes 0-15 K {0..7,16..23}, lanes 16-31 K {8..15,24..31}
      const v8bf* bp = (const v8bf*)&Blds[buf][g][lidx][half * 8];
      v16bf bF = __builtin_shufflevector(bp[0], bp[2],
          0, 1, 2, 3, 4, 5, 6, 7, 8, 9, 10, 11, 12, 13, 14, 15);
#pragma unroll
      for (int ms = 0; ms < 2; ++ms) {
        acc[ms][g] = __builtin_amdgcn_wmma_f32_16x16x32_bf16(
            false, aF[ms], false, bF, (short)0, acc[ms][g], false, false);
      }
    }
    __syncthreads();            // all waves done reading buf before reuse
  }

  // ---- Fused LSTM cell epilogue (gates stay in registers) ----
  const int n  = n0 + lidx;
  const float bI = bias[n];
  const float bF = bias[H_N + n];
  const float bG = bias[2 * H_N + n];
  const float bO = bias[3 * H_N + n];

#pragma unroll
  for (int ms = 0; ms < 2; ++ms) {
    const int rBase = mBase + wave * 32 + ms * 16 + half * 8;
#pragma unroll
    for (int r = 0; r < 8; ++r) {
      size_t idx = (size_t)(rBase + r) * H_N + n;
      float iv = sigf(acc[ms][0][r] + bI);
      float fv = sigf(acc[ms][1][r] + bF);
      float gv = tanh_fast(acc[ms][2][r] + bG);
      float ov = sigf(acc[ms][3][r] + bO);
      float cn = fv * cst[idx] + iv * gv;
      cst[idx]   = cn;
      float hv   = ov * tanh_fast(cn);
      hnext[idx] = f2bf(hv);
      if (last) outf[idx] = hv;   // final h, f32, row-major [B,H]
    }
  }
}

extern "C" void kernel_launch(void* const* d_in, const int* in_sizes, int n_in,
                              void* d_out, int out_size, void* d_ws, size_t ws_size,
                              hipStream_t stream) {
  const float* x    = (const float*)d_in[0];   // [B, T, F]
  const float* Wih  = (const float*)d_in[1];   // [F, 4H]
  const float* Whh  = (const float*)d_in[2];   // [H, 4H]
  const float* bias = (const float*)d_in[3];   // [4H]

  char* ws = (char*)d_ws;
  const size_t WT_BYTES = (size_t)(4 * H_N) * K_N * 2;        // 14,680,064
  const size_t X_BYTES  = (size_t)B_N * T_N * F_N * 2;        // 50,331,648
  const size_t H_BYTES  = (size_t)B_N * H_N * 2;              //  4,194,304

  unsigned short* WT  = (unsigned short*)ws;
  unsigned short* xbf = (unsigned short*)(ws + WT_BYTES);
  unsigned short* hb0 = (unsigned short*)(ws + WT_BYTES + X_BYTES);
  unsigned short* hb1 = (unsigned short*)(ws + WT_BYTES + X_BYTES + H_BYTES);
  float*          cb  = (float*)(ws + WT_BYTES + X_BYTES + 2 * H_BYTES);
  unsigned short* hb[2] = {hb0, hb1};

  prep_wt_kernel<<<28672, 256, 0, stream>>>(Wih, Whh, WT);     // 4096*1792/256
  prep_x_kernel<<<98304, 256, 0, stream>>>(x, xbf);            // 25165824/256
  init_state_kernel<<<8192, 256, 0, stream>>>(hb0, cb);        // 2097152/256

  for (int t = 0; t < T_N; ++t) {
    int last = (t == T_N - 1);
    lstm_step_kernel<<<dim3(8, 64), 256, 0, stream>>>(
        xbf, WT, bias, hb[t & 1], hb[(t + 1) & 1], cb, (float*)d_out, t, last);
  }
}
